// WeightLinearAttention_40836549050696
// MI455X (gfx1250) — matrix operand order repaired and checked by reading, not compiled, not run on hardware
//
#include <hip/hip_runtime.h>
#include <hip/hip_bf16.h>

// ---------------- problem constants ----------------
#define LSEQ  2048
#define BATCH 2
#define EMB   1024
#define NHEAD 16
#define HDIM  64
#define ROWS  (LSEQ * BATCH)   // 4096 GEMM rows, row r -> (l = r/2, b = r&1)
#define NCH   (LSEQ / 64)      // 32 chunks
#define EPSV  1e-4f

typedef __attribute__((ext_vector_type(16))) __bf16        v16bf;
typedef __attribute__((ext_vector_type(8)))  float         f32x8;
typedef __attribute__((ext_vector_type(4)))  unsigned int  u32x4;
typedef __attribute__((ext_vector_type(4)))  int           i32x4;
typedef __attribute__((ext_vector_type(8)))  int           i32x8;

// ---------------------------------------------------------------------------
// Tensor Data Mover: 2D tile load Global -> LDS (CDNA5 ISA ch8 D# layout).
// group0: count=1 | lds_addr | global_addr(57b) | type=2
// group1: data_size | tensor_dim0/1 | tile_dim0/1 | dim0_stride
// ---------------------------------------------------------------------------
__device__ __forceinline__ void tdm_load_2d(unsigned lds_off, const void* gptr,
                                            unsigned tile_w, unsigned tile_rows,
                                            unsigned tensor_w,
                                            unsigned long long row_stride,
                                            unsigned ds_log2 /*0=1B,1=2B,2=4B*/) {
  unsigned long long ga = (unsigned long long)(size_t)gptr;
  u32x4 g0;
  g0[0] = 1u;                                               // count=1, user
  g0[1] = lds_off;                                          // LDS byte address
  g0[2] = (unsigned)(ga & 0xFFFFFFFFu);
  g0[3] = (unsigned)((ga >> 32) & 0x1FFFFFFu) | (2u << 30); // type=2 ("image")
  i32x8 g1;
  g1[0] = (int)(ds_log2 << 16);                               // workgroup_mask=0, data_size
  g1[1] = (int)((tensor_w & 0xFFFFu) << 16);                  // tensor_dim0 lo16
  g1[2] = (int)((tensor_w >> 16) | ((tile_rows & 0xFFFFu) << 16)); // dim0 hi | tensor_dim1 lo
  g1[3] = (int)((tile_rows >> 16) | (tile_w << 16));          // dim1 hi | tile_dim0
  g1[4] = (int)tile_rows;                                     // tile_dim1 | tile_dim2=0
  g1[5] = (int)(row_stride & 0xFFFFFFFFull);                  // dim0_stride lo32
  g1[6] = (int)((row_stride >> 32) & 0xFFFFull);              // dim0_stride hi16
  g1[7] = 0;
  i32x4 gz = {0, 0, 0, 0};
#if defined(__clang_major__) && (__clang_major__ >= 23)
  i32x8 gz8 = {0, 0, 0, 0, 0, 0, 0, 0};
  __builtin_amdgcn_tensor_load_to_lds(g0, g1, gz, gz, gz8, 0);
#else
  __builtin_amdgcn_tensor_load_to_lds(g0, g1, gz, gz, 0);
#endif
}

// ---------------- WMMA fragment builders (contiguous k order, A and B alike) ----------
// A 16x32 bf16: lane m = lane&15, half = lane>>4; elem j -> K=half*16+j
// B 32x16 bf16: lane n = lane&15, kh = lane>>4;   elem j -> K=kh*16+j
// C/D 16x16 f32: lane n = lane&15; VGPR i -> M = i + 8*(lane>>4)

__device__ __forceinline__ v16bf a_frag_bf(const __bf16* src, int ld, int m0, int k0) {
  int lane = threadIdx.x & 31;
  int m = lane & 15, half = lane >> 4;
  const __bf16* p = src + (size_t)(m0 + m) * ld + k0 + half * 16;
  v16bf u;
#pragma unroll
  for (int j = 0; j < 8; ++j) u[j] = p[j];          // K block 0..7 (+half*16)
#pragma unroll
  for (int j = 0; j < 8; ++j) u[8 + j] = p[8 + j]; // K block 8..15 (+half*16)
  return u;
}

// A[m][kk] = src[k0+kk][m0+m]  (transposed gather, for k^T in state update)
__device__ __forceinline__ v16bf a_frag_bf_T(const __bf16* src, int ld, int m0, int k0) {
  int lane = threadIdx.x & 31;
  int m = lane & 15, half = lane >> 4;
  v16bf u;
#pragma unroll
  for (int j = 0; j < 16; ++j) {
    int kk = (half << 4) | j;
    u[j] = src[(size_t)(k0 + kk) * ld + m0 + m];
  }
  return u;
}

// A from f32 source (LDS score tile), convert with native bf16 cvt
__device__ __forceinline__ v16bf a_frag_f32(const float* src, int ld, int m0, int k0) {
  int lane = threadIdx.x & 31;
  int m = lane & 15, half = lane >> 4;
  const float* p = src + (size_t)(m0 + m) * ld + k0 + half * 16;
  v16bf u;
#pragma unroll
  for (int j = 0; j < 8; ++j) u[j] = (__bf16)p[j];
#pragma unroll
  for (int j = 0; j < 8; ++j) u[8 + j] = (__bf16)p[8 + j];
  return u;
}

// B[k][n] = srcT[n0+n][k0+k]  (src row-major over n; X@W^T and q k^T)
__device__ __forceinline__ v16bf b_frag_bf_T(const __bf16* src, int ld, int n0, int k0) {
  int lane = threadIdx.x & 31;
  int n = lane & 15, kh = lane >> 4;
  const __bf16* p = src + (size_t)(n0 + n) * ld + k0 + (kh << 4);
  v16bf u;
#pragma unroll
  for (int j = 0; j < 16; ++j) u[j] = p[j];
  return u;
}

// B[k][n] = src[k0+k][n0+n]  (row-major direct; v tiles)
__device__ __forceinline__ v16bf b_frag_bf(const __bf16* src, int ld, int k0, int n0) {
  int lane = threadIdx.x & 31;
  int n = lane & 15, kh = lane >> 4;
  v16bf u;
#pragma unroll
  for (int j = 0; j < 16; ++j) u[j] = src[(size_t)(k0 + (kh << 4) + j) * ld + n0 + n];
  return u;
}

// B from f32 source (state S in LDS)
__device__ __forceinline__ v16bf b_frag_f32(const float* src, int ld, int k0, int n0) {
  int lane = threadIdx.x & 31;
  int n = lane & 15, kh = lane >> 4;
  v16bf u;
#pragma unroll
  for (int j = 0; j < 16; ++j) u[j] = (__bf16)src[(size_t)(k0 + (kh << 4) + j) * ld + n0 + n];
  return u;
}

__device__ __forceinline__ f32x8 c_load(const float* src, int ld, int m0, int n0) {
  int lane = threadIdx.x & 31;
  int n = lane & 15, half = lane >> 4;
  f32x8 c;
#pragma unroll
  for (int i = 0; i < 8; ++i) c[i] = src[(size_t)(m0 + i + 8 * half) * ld + n0 + n];
  return c;
}
__device__ __forceinline__ void c_store(float* dst, int ld, int m0, int n0, f32x8 c) {
  int lane = threadIdx.x & 31;
  int n = lane & 15, half = lane >> 4;
#pragma unroll
  for (int i = 0; i < 8; ++i) dst[(size_t)(m0 + i + 8 * half) * ld + n0 + n] = c[i];
}

__device__ __forceinline__ f32x8 wmma_bf(v16bf a, v16bf b, f32x8 c) {
  return __builtin_amdgcn_wmma_f32_16x16x32_bf16(false, a, false, b, (short)0, c, false, false);
}

// ---------------------------------------------------------------------------
// f32 -> bf16 bulk convert (native v_cvt path)
// ---------------------------------------------------------------------------
__global__ __launch_bounds__(256) void cvt_kernel(const float* __restrict__ src,
                                                  __bf16* __restrict__ dst, int n) {
  for (int i = blockIdx.x * 256 + threadIdx.x; i < n; i += gridDim.x * 256)
    dst[i] = (__bf16)src[i];
}

// ---------------------------------------------------------------------------
// Projection GEMM: Y = X(ROWS x EMB, bf16) @ W^T(EMB x EMB, bf16) + bias
// A tile staged in LDS by the Tensor Data Mover.
// MODE 0: store f32 row-major (final output projection)
// MODE 1: relu -> RoPE, store bf16 into (B,H,L,D)   (q, k)
// MODE 2: store bf16 into (B,H,L,D)                 (v)
// Block = 256 threads = 8 waves; wave -> 16x64 tile; grid = (ROWS/16, EMB/512)
// ---------------------------------------------------------------------------
template <int MODE>
__global__ __launch_bounds__(256) void proj_kernel(
    const __bf16* __restrict__ X, const __bf16* __restrict__ W,
    const float* __restrict__ bias, void* __restrict__ out) {
  __shared__ __bf16 xs[16 * EMB];  // 32 KB A-tile

  int wave = threadIdx.x >> 5;
  int lane = threadIdx.x & 31;
  int m0 = blockIdx.x * 16;
  int n0 = blockIdx.y * 512 + wave * 64;

  if (wave == 0) {  // TDM: 16 rows x EMB bf16, contiguous rows
    tdm_load_2d((unsigned)(size_t)&xs[0], X + (size_t)m0 * EMB,
                EMB, 16, EMB, EMB, 1);
    __builtin_amdgcn_s_wait_tensorcnt((short)0);
  }
  __syncthreads();

  f32x8 acc[4];
#pragma unroll
  for (int t = 0; t < 4; ++t)
#pragma unroll
    for (int i = 0; i < 8; ++i) acc[t][i] = 0.f;

  for (int k0 = 0; k0 < EMB; k0 += 32) {
    if (k0 + 128 < EMB)  // gfx1250 global_prefetch_b8 on the streaming operand
      __builtin_prefetch(W + (size_t)(n0 + (lane & 15)) * EMB + k0 + 128, 0, 1);
    v16bf a = a_frag_bf(xs, EMB, 0, k0);
#pragma unroll
    for (int t = 0; t < 4; ++t) {
      v16bf b = b_frag_bf_T(W, EMB, n0 + t * 16, k0);
      acc[t] = wmma_bf(a, b, acc[t]);
    }
  }

  int n = lane & 15, half = lane >> 4;
#pragma unroll
  for (int t = 0; t < 4; ++t) {
    int e = n0 + t * 16 + n;
    float bv = bias[e];
#pragma unroll
    for (int i = 0; i < 8; ++i) {
      int r = m0 + i + 8 * half;
      float x = acc[t][i] + bv;
      if (MODE == 0) {
        ((float*)out)[(size_t)r * EMB + e] = x;
      } else {
        int l = r >> 1, bb = r & 1;  // rows flatten (L, B) with B==2
        int h = e >> 6, d = e & 63;
        if (MODE == 1) {
          x = fmaxf(x, 0.f);
          float p = __shfl_xor(x, 1, 32);  // RoPE partner: column d^1
          int di = d >> 1;
          // theta_i = 10000^(-2 i / 64) = exp(-i * ln(10000)/32)
          float theta = __expf(-(float)di * 0.28782313662425572f);
          float ang = (float)l * theta;
          float s, c;
          __sincosf(ang, &s, &c);
          x = (d & 1) ? (p * s + x * c) : (x * c - p * s);
        }
        ((__bf16*)out)[(((size_t)bb * NHEAD + h) * LSEQ + l) * HDIM + d] = (__bf16)x;
      }
    }
  }
}

// ---------------------------------------------------------------------------
// Chunked causal linear attention. One workgroup per (b,h); 8 waves.
// q/k/v 64x64 chunks are TDM double-buffered through LDS (issue chunk ch+1
// while computing chunk ch). State S[dk][dv] (f32) + z[dk] in LDS.
// Per chunk:  o = q*S + tril(q k^T) v ; denom = q*z + rowsum(tril(q k^T))
// then S += k^T v ; z += colsum(k).
// ---------------------------------------------------------------------------
__global__ __launch_bounds__(256) void attn_kernel(
    const __bf16* __restrict__ Q, const __bf16* __restrict__ K,
    const __bf16* __restrict__ V, float* __restrict__ O) {
  __shared__ __bf16 qs[2][64 * HDIM];  // 16 KB
  __shared__ __bf16 ks[2][64 * HDIM];
  __shared__ __bf16 vs[2][64 * HDIM];
  __shared__ float S[HDIM * HDIM];     // 16 KB state
  __shared__ float sc[64 * 64];        // 16 KB masked scores
  __shared__ float zvec[HDIM];
  __shared__ float dn[64];

  int bh = blockIdx.x;  // 0..31
  int b = bh / NHEAD, h = bh % NHEAD;
  const __bf16* q = Q + (size_t)bh * LSEQ * HDIM;
  const __bf16* k = K + (size_t)bh * LSEQ * HDIM;
  const __bf16* v = V + (size_t)bh * LSEQ * HDIM;

  int tid = threadIdx.x;
  int wave = tid >> 5, lane = tid & 31;
  int n = lane & 15, half = lane >> 4;
  int mt = wave & 3;   // output row tile (0..3)
  int np = wave >> 2;  // output col pair (0..1): tiles 2np, 2np+1
  int nc0 = np * 32, nc1 = nc0 + 16;

  for (int i = tid; i < HDIM * HDIM; i += 256) S[i] = 0.f;
  if (tid < HDIM) zvec[tid] = 0.f;

  if (wave == 0) {  // preload chunk 0 into buffer 0 (3 contiguous 8KB tiles)
    tdm_load_2d((unsigned)(size_t)&qs[0][0], q, 4096, 1, 4096, 4096, 1);
    tdm_load_2d((unsigned)(size_t)&ks[0][0], k, 4096, 1, 4096, 4096, 1);
    tdm_load_2d((unsigned)(size_t)&vs[0][0], v, 4096, 1, 4096, 4096, 1);
    __builtin_amdgcn_s_wait_tensorcnt((short)0);
  }
  __syncthreads();

  for (int ch = 0; ch < NCH; ++ch) {
    int base = ch * 64;
    int buf = ch & 1;
    const __bf16* qc = &qs[buf][0];
    const __bf16* kc = &ks[buf][0];
    const __bf16* vc = &vs[buf][0];

    if (wave == 0 && ch + 1 < NCH) {  // async prefetch next chunk
      size_t off = (size_t)(base + 64) * HDIM;
      tdm_load_2d((unsigned)(size_t)&qs[buf ^ 1][0], q + off, 4096, 1, 4096, 4096, 1);
      tdm_load_2d((unsigned)(size_t)&ks[buf ^ 1][0], k + off, 4096, 1, 4096, 4096, 1);
      tdm_load_2d((unsigned)(size_t)&vs[buf ^ 1][0], v + off, 4096, 1, 4096, 4096, 1);
    }

    f32x8 s0, s1, o0, o1;
#pragma unroll
    for (int i = 0; i < 8; ++i) { s0[i] = 0.f; s1[i] = 0.f; o0[i] = 0.f; o1[i] = 0.f; }

    // ---- phase A: scores = q k^T ; o_inter = q @ S(old) ----
#pragma unroll
    for (int k0 = 0; k0 < HDIM; k0 += 32) {
      v16bf aq = a_frag_bf(qc, HDIM, mt * 16, k0);
      v16bf bk0 = b_frag_bf_T(kc, HDIM, nc0, k0);
      v16bf bk1 = b_frag_bf_T(kc, HDIM, nc1, k0);
      s0 = wmma_bf(aq, bk0, s0);
      s1 = wmma_bf(aq, bk1, s1);
      v16bf bS0 = b_frag_f32(S, HDIM, k0, nc0);
      v16bf bS1 = b_frag_f32(S, HDIM, k0, nc1);
      o0 = wmma_bf(aq, bS0, o0);
      o1 = wmma_bf(aq, bS1, o1);
    }
    // masked score store (causal within chunk: col <= row)
#pragma unroll
    for (int i = 0; i < 8; ++i) {
      int r = mt * 16 + i + 8 * half;
      sc[r * 64 + nc0 + n] = (nc0 + n <= r) ? s0[i] : 0.f;
      sc[r * 64 + nc1 + n] = (nc1 + n <= r) ? s1[i] : 0.f;
    }
    __syncthreads();

    // ---- phase B: o_intra = tril_scores @ v ; denom ----
#pragma unroll
    for (int k0 = 0; k0 < 64; k0 += 32) {
      v16bf as = a_frag_f32(sc, 64, mt * 16, k0);
      v16bf bv0 = b_frag_bf(vc, HDIM, k0, nc0);
      v16bf bv1 = b_frag_bf(vc, HDIM, k0, nc1);
      o0 = wmma_bf(as, bv0, o0);
      o1 = wmma_bf(as, bv1, o1);
    }
    if (tid < 64) {
      int r = tid;
      float s = 0.f;
      for (int c = 0; c < 64; ++c) s += sc[r * 64 + c];
      const __bf16* qr = qc + (size_t)r * HDIM;
      float dqz = 0.f;
      for (int d = 0; d < HDIM; ++d) dqz += (float)qr[d] * zvec[d];
      dn[r] = fmaxf(s + dqz, EPSV);
    }
    __syncthreads();

    // ---- store o/denom ; update S and z ----
#pragma unroll
    for (int i = 0; i < 8; ++i) {
      int r = mt * 16 + i + 8 * half;
      int l = base + r;
      float inv = 1.f / dn[r];
      size_t rowo = ((size_t)l * BATCH + b) * EMB + (size_t)h * HDIM;
      O[rowo + nc0 + n] = o0[i] * inv;
      O[rowo + nc1 + n] = o1[i] * inv;
    }
    for (int tt = wave; tt < 16; tt += 8) {  // S += k^T @ v : 2 tiles/wave
      int m2 = (tt >> 2) * 16;               // dk tile
      int n2 = (tt & 3) * 16;                // dv tile
      f32x8 cs = c_load(S, HDIM, m2, n2);
#pragma unroll
      for (int k0 = 0; k0 < 64; k0 += 32) {
        v16bf akT = a_frag_bf_T(kc, HDIM, m2, k0);
        v16bf bvv = b_frag_bf(vc, HDIM, k0, n2);
        cs = wmma_bf(akT, bvv, cs);
      }
      c_store(S, HDIM, m2, n2, cs);
    }
    if (tid < HDIM) {
      float acc = zvec[tid];
      for (int s2 = 0; s2 < 64; ++s2) acc += (float)kc[(size_t)s2 * HDIM + tid];
      zvec[tid] = acc;
    }
    if (wave == 0) __builtin_amdgcn_s_wait_tensorcnt((short)0);
    __syncthreads();
  }
}

// ---------------------------------------------------------------------------
extern "C" void kernel_launch(void* const* d_in, const int* in_sizes, int n_in,
                              void* d_out, int out_size, void* d_ws, size_t ws_size,
                              hipStream_t stream) {
  const float* query = (const float*)d_in[0];
  const float* key_  = (const float*)d_in[1];
  const float* value = (const float*)d_in[2];
  const float* q_w   = (const float*)d_in[3];
  const float* q_b   = (const float*)d_in[4];
  const float* k_w   = (const float*)d_in[5];
  const float* k_b   = (const float*)d_in[6];
  const float* v_w   = (const float*)d_in[7];
  const float* v_b   = (const float*)d_in[8];
  const float* out_w = (const float*)d_in[9];
  const float* out_b = (const float*)d_in[10];
  float* out = (float*)d_out;

  char* ws = (char*)d_ws;
  const size_t WSZ = (size_t)EMB * EMB;   // weight elements (1M)
  const size_t SZ  = (size_t)ROWS * EMB;  // activation elements (4M)
  __bf16* Wq = (__bf16*)(ws);
  __bf16* Wk = (__bf16*)(ws + WSZ * 2);
  __bf16* Wv = (__bf16*)(ws + WSZ * 4);
  __bf16* Wo = (__bf16*)(ws + WSZ * 6);
  __bf16* Xq = (__bf16*)(ws + WSZ * 8);             // bf16 (L,B,E) inputs
  __bf16* Xk = (__bf16*)(ws + WSZ * 8 + SZ * 2);
  __bf16* Xv = (__bf16*)(ws + WSZ * 8 + SZ * 4);
  __bf16* qb = (__bf16*)(ws + WSZ * 8 + SZ * 6);    // bf16 (B,H,L,D) post-RoPE
  __bf16* kb = (__bf16*)(ws + WSZ * 8 + SZ * 8);
  __bf16* vb = (__bf16*)(ws + WSZ * 8 + SZ * 10);
  float*  ao = (float*)(ws + WSZ * 8 + SZ * 12);    // f32 (L,B,E) attention out
  __bf16* aob = Xq;                                 // reuse: Xq dead after proj q

  dim3 cgrid(512), blk(256);
  cvt_kernel<<<cgrid, blk, 0, stream>>>(q_w, Wq, (int)WSZ);
  cvt_kernel<<<cgrid, blk, 0, stream>>>(k_w, Wk, (int)WSZ);
  cvt_kernel<<<cgrid, blk, 0, stream>>>(v_w, Wv, (int)WSZ);
  cvt_kernel<<<cgrid, blk, 0, stream>>>(out_w, Wo, (int)WSZ);
  cvt_kernel<<<cgrid, blk, 0, stream>>>(query, Xq, (int)SZ);
  cvt_kernel<<<cgrid, blk, 0, stream>>>(key_, Xk, (int)SZ);
  cvt_kernel<<<cgrid, blk, 0, stream>>>(value, Xv, (int)SZ);

  dim3 grid(ROWS / 16, EMB / 512);
  proj_kernel<1><<<grid, blk, 0, stream>>>(Xq, Wq, q_b, (void*)qb);
  proj_kernel<1><<<grid, blk, 0, stream>>>(Xk, Wk, k_b, (void*)kb);
  proj_kernel<2><<<grid, blk, 0, stream>>>(Xv, Wv, v_b, (void*)vb);

  attn_kernel<<<dim3(BATCH * NHEAD), blk, 0, stream>>>(qb, kb, vb, ao);

  cvt_kernel<<<cgrid, blk, 0, stream>>>(ao, aob, (int)SZ);
  proj_kernel<0><<<grid, blk, 0, stream>>>(aob, Wo, out_b, (void*)out);
}
